// EFIN_59554016526353
// MI455X (gfx1250) — matrix operand order repaired and checked by reading, not compiled, not run on hardware
//
// MI455X (gfx1250) uplift-model forward pass.
// bf16 activations + v_wmma_f32_16x16x32_bf16 (f32 accum) for all GEMMs;
// fused per-sample attention in LDS; BatchNorm folded to scale/shift after a
// batch-stats reduction; gfx1250 async global->LDS copies (ASYNCcnt) for the
// GEMM A-staging (double buffered) and the embedding gather.
#include <hip/hip_runtime.h>
#include <hip/hip_bf16.h>

#define B_N 16384
#define L_N 50
#define E_N 64

typedef __attribute__((ext_vector_type(16))) __bf16 v16bf;
typedef __attribute__((ext_vector_type(8)))  float  v8f;
typedef unsigned short u16;
typedef unsigned int   u32;

union Frag16 { v16bf v; u32 u[8]; };

__device__ __forceinline__ u16 f2bf(float x) {
  u32 u = __float_as_uint(x);
  u32 r = u + 0x7FFFu + ((u >> 16) & 1u);   // round-to-nearest-even
  return (u16)(r >> 16);
}
__device__ __forceinline__ float sigmoidf_(float x) { return 1.0f / (1.0f + __expf(-x)); }

// ---- gfx1250 async global->LDS copy (tracked by ASYNCcnt) -------------------
// 16B per lane; both addresses must be 16B aligned.
__device__ __forceinline__ void async_copy16(const void* g, u32 ldsOff) {
  asm volatile("global_load_async_to_lds_b128 %0, %1, off"
               :: "v"(ldsOff), "v"((unsigned long long)(size_t)g)
               : "memory");
}
__device__ __forceinline__ void wait_async0() {
  asm volatile("s_wait_asynccnt 0x0" ::: "memory");
}
__device__ __forceinline__ void wait_async1() {
  asm volatile("s_wait_asynccnt 0x1" ::: "memory");
}

// A-matrix fragment (16x32 bf16, M rows in lanes, contiguous K: k = 16*half + s).
__device__ __forceinline__ v16bf load_fragA(const u16* base, int stride, int row0, int k0, int lane) {
  const u16* r = base + (size_t)(row0 + (lane & 15)) * stride + k0;
  const int half8 = ((lane >> 4) & 1) * 8;
  Frag16 f;
#pragma unroll
  for (int v = 0; v < 8; ++v) {
    int k = half8 * 2 + v * 2;                          // {0,16} + 2*v
    f.u[v] = *(const u32*)(r + k);
  }
  return f.v;
}
// B fragment from row-major [N][K] storage (torch weights W[n][k]; B[k][n]=W[n][k]).
__device__ __forceinline__ v16bf load_fragB_nk(const u16* base, int stride, int n0, int k0, int lane) {
  const u16* r = base + (size_t)(n0 + (lane & 15)) * stride + k0 + ((lane >> 4) & 1) * 16;
  Frag16 f;
#pragma unroll
  for (int v = 0; v < 8; ++v) f.u[v] = *(const u32*)(r + v * 2);
  return f.v;
}
// B fragment from row-major [K][N] storage (e.g. V matrix: B[k][n] = V[k][n]).
__device__ __forceinline__ v16bf load_fragB_kn(const u16* base, int stride, int k0, int n0, int lane) {
  const u16* c = base + (size_t)(k0 + ((lane >> 4) & 1) * 16) * stride + n0 + (lane & 15);
  Frag16 f;
#pragma unroll
  for (int v = 0; v < 8; ++v) {
    u32 lo = c[(2 * v) * stride];
    u32 hi = c[(2 * v + 1) * stride];
    f.u[v] = (lo & 0xFFFFu) | (hi << 16);
  }
  return f.v;
}
__device__ __forceinline__ v8f wmma_bf16(v16bf a, v16bf b, v8f c) {
  return __builtin_amdgcn_wmma_f32_16x16x32_bf16(false, a, false, b, (short)0, c, false, false);
}

// ---------------- f32 -> bf16 weight conversion ----------------
__global__ __launch_bounds__(256) void cvt_bf16_kernel(const float* __restrict__ src,
                                                       u16* __restrict__ dst, int n) {
  int i = blockIdx.x * 256 + threadIdx.x;
  if (i < n) dst[i] = f2bf(src[i]);
}

// ---------------- fused per-sample attention + treatment path ----------------
// One workgroup (256 thr = 8 waves) per sample. LDS ~63 KB.
__global__ __launch_bounds__(256) void attn_kernel(
    const int* __restrict__ xin, const float* __restrict__ rebate,
    const float* __restrict__ emb,
    const float* __restrict__ Qb, const float* __restrict__ Kb, const float* __restrict__ Vb,
    const u16* __restrict__ qwb, const u16* __restrict__ kwb, const u16* __restrict__ vwb,
    const float* __restrict__ a1w, const float* __restrict__ a2w, const float* __restrict__ a2b,
    const float* __restrict__ a3w, const float* __restrict__ tw, const float* __restrict__ tb,
    u16* __restrict__ flat, u16* __restrict__ tfeat) {
  __shared__ __align__(16) float sXrep[L_N][E_N];   // unnormalized embeddings (t-path)
  __shared__ __align__(16) u16 sXr[64][E_N];        // normalized bf16, rows >=50 zero
  __shared__ __align__(16) u16 sQ[64][E_N];         // Q, later reused for attn (bf16)
  __shared__ __align__(16) u16 sK[64][E_N];
  __shared__ __align__(16) u16 sV[64][E_N];
  __shared__ float sS[64][64];                      // sigmoid(scores); later s2
  __shared__ float sNorm[64], sT[64], sS1[64], sAtt[64];

  const int b = blockIdx.x, tid = threadIdx.x;
  const int lane = tid & 31, wave = tid >> 5;

  // 1) gather embedding rows via async global->LDS (16B granules); t_rep
  for (int i = tid; i < L_N * (E_N / 4); i += 256) {        // 800 x 16B
    int l = i >> 4, e4 = (i & 15) << 2;
    const float* g = emb + (size_t)xin[b * L_N + l] * E_N + e4;
    async_copy16(g, (u32)(size_t)&sXrep[l][e4]);
  }
  if (tid < 64) sT[tid] = rebate[b] * tw[tid] + tb[tid];
  wait_async0();
  __syncthreads();

  // 2) column norms over L (dim=1 of [B,L,E])
  if (tid < 64) {
    float s = 0.f;
    for (int l = 0; l < L_N; ++l) { float v = sXrep[l][tid]; s += v * v; }
    sNorm[tid] = 1.0f / sqrtf(s);
  }
  __syncthreads();

  // 3) normalize -> bf16, zero-pad rows 50..63; s1 = sigmoid(t_rep @ a1w.T)
  for (int i = tid; i < 64 * 64; i += 256) {
    int l = i >> 6, e = i & 63;
    sXr[l][e] = f2bf((l < L_N) ? sXrep[l][e] * sNorm[e] : 0.f);
  }
  if (tid < 64) {
    float s = 0.f;
    const float* row = a1w + tid * 64;
    for (int k = 0; k < 64; ++k) s += sT[k] * row[k];
    sS1[tid] = sigmoidf_(s);
  }
  __syncthreads();

  // 4) Q,K,V = xr @ W.T + b via WMMA (48 16x16 tiles over 8 waves)
  for (int t = wave; t < 48; t += 8) {
    int mat = t >> 4, tile = t & 15, mt = tile >> 2, nt = tile & 3;
    const u16* W = (mat == 0) ? qwb : (mat == 1) ? kwb : vwb;
    const float* bias = (mat == 0) ? Qb : (mat == 1) ? Kb : Vb;
    u16(*dst)[E_N] = (mat == 0) ? sQ : (mat == 1) ? sK : sV;
    v8f acc = {};
#pragma unroll
    for (int k0 = 0; k0 < 64; k0 += 32) {
      v16bf a = load_fragA(&sXr[0][0], 64, mt * 16, k0, lane);
      v16bf w = load_fragB_nk(W, 64, nt * 16, k0, lane);
      acc = wmma_bf16(a, w, acc);
    }
    int n = nt * 16 + (lane & 15);
    float bn = bias[n];
    int mbase = mt * 16 + ((lane >> 4) ? 8 : 0);
#pragma unroll
    for (int r = 0; r < 8; ++r) dst[mbase + r][n] = f2bf(acc[r] + bn);
  }
  __syncthreads();

  // 5) scores = QK^T/8, then sigmoid (B[e][m]=K[m][e] -> nk loader on sK)
  for (int t = wave; t < 16; t += 8) {
    int mt = t >> 2, nt = t & 3;
    v8f acc = {};
#pragma unroll
    for (int k0 = 0; k0 < 64; k0 += 32) {
      v16bf a = load_fragA(&sQ[0][0], 64, mt * 16, k0, lane);
      v16bf bk = load_fragB_nk(&sK[0][0], 64, nt * 16, k0, lane);
      acc = wmma_bf16(a, bk, acc);
    }
    int n = nt * 16 + (lane & 15);
    int mbase = mt * 16 + ((lane >> 4) ? 8 : 0);
#pragma unroll
    for (int r = 0; r < 8; ++r) sS[mbase + r][n] = sigmoidf_(acc[r] * 0.125f);
  }
  __syncthreads();

  // 6) softmax over dim=1 (over l, per column m); write attn as bf16 into sQ
  if (tid < 64) {
    int m = tid;
    if (m < L_N) {
      float mx = -1e30f;
      for (int l = 0; l < L_N; ++l) mx = fmaxf(mx, sS[l][m]);
      float sum = 0.f;
      for (int l = 0; l < L_N; ++l) { float e = __expf(sS[l][m] - mx); sS[l][m] = e; sum += e; }
      float inv = 1.f / sum;
      for (int l = 0; l < L_N; ++l) sQ[l][m] = f2bf(sS[l][m] * inv);
    } else {
      for (int l = 0; l < L_N; ++l) sQ[l][m] = 0;   // pad columns -> zero weight
    }
  }
  for (int i = tid; i < (64 - L_N) * 64; i += 256) sQ[L_N + (i >> 6)][i & 63] = 0;
  __syncthreads();

  // 7a) sa_out = attn @ V -> flat (bf16, global).  7b) s2 into sS (reuse).
  for (int t = wave; t < 16; t += 8) {
    int mt = t >> 2, nt = t & 3;
    v8f acc = {};
#pragma unroll
    for (int k0 = 0; k0 < 64; k0 += 32) {
      v16bf a = load_fragA(&sQ[0][0], 64, mt * 16, k0, lane);
      v16bf bv = load_fragB_kn(&sV[0][0], 64, k0, nt * 16, lane);
      acc = wmma_bf16(a, bv, acc);
    }
    int n = nt * 16 + (lane & 15);
    int mbase = mt * 16 + ((lane >> 4) ? 8 : 0);
    u16* outp = flat + (size_t)b * (L_N * E_N);
#pragma unroll
    for (int r = 0; r < 8; ++r) {
      int l = mbase + r;
      if (l < L_N) outp[l * E_N + n] = f2bf(acc[r]);
    }
  }
  for (int i = tid; i < L_N * 64; i += 256) {        // s2 = sigmoid(x_rep@a2w.T + a2b)
    int l = i >> 6, e = i & 63;
    const float* row = a2w + e * 64;
    float d = a2b[e];
    for (int k = 0; k < 64; ++k) d += sXrep[l][k] * row[k];
    sS[l][e] = sigmoidf_(d);
  }
  __syncthreads();

  // 8) att logits, softmax over l, t_feat
  if (tid < L_N) {
    float s = 0.f;
    for (int e = 0; e < 64; ++e) s += fmaxf(sS1[e] + sS[tid][e], 0.f) * a3w[e];
    sAtt[tid] = s;
  }
  __syncthreads();
  if (tid == 0) {
    float mx = -1e30f;
    for (int l = 0; l < L_N; ++l) mx = fmaxf(mx, sAtt[l]);
    float sum = 0.f;
    for (int l = 0; l < L_N; ++l) { float e = __expf(sAtt[l] - mx); sAtt[l] = e; sum += e; }
    float inv = 1.f / sum;
    for (int l = 0; l < L_N; ++l) sAtt[l] *= inv;
  }
  __syncthreads();
  if (tid < 64) {
    float f = 0.f;
    for (int l = 0; l < L_N; ++l) f += sAtt[l] * sXrep[l][tid];
    tfeat[(size_t)b * 64 + tid] = f2bf(f);
  }
}

// ---- generic bf16 WMMA GEMM: out[M,N] = A[M,K] @ W[N,K]^T + bias ----
// Double-buffered async global->LDS staging of A (ASYNCcnt-tracked).
__global__ __launch_bounds__(256) void gemm_bf16_kernel(
    const u16* __restrict__ A, const u16* __restrict__ W, const float* __restrict__ bias,
    float* __restrict__ out, int K, int N) {
  __shared__ __align__(16) u16 sA[2][64][32];
  const int tid = threadIdx.x, lane = tid & 31, wave = tid >> 5;
  const int row0 = blockIdx.x * 64;
  const int mt = wave & 3, nt0 = wave >> 2, NT = N >> 4;
  v8f acc[2] = {};
  const int r = tid >> 2, c = (tid & 3) * 8;        // cooperative A staging map
  const u16* gbase = A + (size_t)(row0 + r) * K + c;

  async_copy16(gbase, (u32)(size_t)&sA[0][r][c]);   // prologue: slab 0
  for (int k0 = 0; k0 < K; k0 += 32) {
    const int buf = (k0 >> 5) & 1;
    const bool more = (k0 + 32 < K);                // uniform
    if (more) {
      // prefetch next slab into the buffer fenced by the previous end-barrier
      async_copy16(gbase + k0 + 32, (u32)(size_t)&sA[buf ^ 1][r][c]);
      wait_async1();                                // own slab k0 done (in-order)
    } else {
      wait_async0();
    }
    __syncthreads();                                // all waves' slab visible
    v16bf a = load_fragA(&sA[buf][0][0], 32, mt * 16, 0, lane);
#pragma unroll
    for (int j = 0; j < 2; ++j) {
      int nt = nt0 + j * 2;
      if (nt < NT) {
        v16bf w = load_fragB_nk(W, K, nt * 16, k0, lane);
        acc[j] = wmma_bf16(a, w, acc[j]);
      }
    }
    __syncthreads();                                // readers done before reuse
  }
  const int n_ = lane & 15;
  const int mbase = row0 + mt * 16 + ((lane >> 4) ? 8 : 0);
#pragma unroll
  for (int j = 0; j < 2; ++j) {
    int nt = nt0 + j * 2;
    if (nt < NT) {
      int n = nt * 16 + n_;
      float bn = bias[n];
#pragma unroll
      for (int rr = 0; rr < 8; ++rr) out[(size_t)(mbase + rr) * N + n] = acc[j][rr] + bn;
    }
  }
}

// ---------------- BatchNorm (training): batch stats -> per-feature scale/shift -------
__global__ __launch_bounds__(256) void bnstats_kernel(
    const float* __restrict__ X, int M, int C,
    const float* __restrict__ gamma, const float* __restrict__ beta,
    float* __restrict__ scale, float* __restrict__ shift) {
  __shared__ float red[512];
  const int cidx = blockIdx.x;
  float s = 0.f, sq = 0.f;
  for (int i = threadIdx.x; i < M; i += 256) {
    float v = X[(size_t)i * C + cidx];
    s += v; sq += v * v;
  }
  red[threadIdx.x] = s; red[256 + threadIdx.x] = sq;
  __syncthreads();
  for (int o = 128; o > 0; o >>= 1) {
    if (threadIdx.x < o) {
      red[threadIdx.x] += red[threadIdx.x + o];
      red[256 + threadIdx.x] += red[256 + threadIdx.x + o];
    }
    __syncthreads();
  }
  if (threadIdx.x == 0) {
    float mean = red[0] / (float)M;
    float var = red[256] / (float)M - mean * mean;       // biased variance
    float a = gamma[cidx] / sqrtf(var + 1e-5f);
    scale[cidx] = a;
    shift[cidx] = beta[cidx] - a * mean;
  }
}

__global__ __launch_bounds__(256) void bnrelu_bf16_kernel(
    const float* __restrict__ X, const float* __restrict__ scale,
    const float* __restrict__ shift, u16* __restrict__ Y, int n, int Cmask) {
  int i = blockIdx.x * 256 + threadIdx.x;
  if (i < n) {
    int cidx = i & Cmask;
    Y[i] = f2bf(fmaxf(scale[cidx] * X[i] + shift[cidx], 0.f));
  }
}

// ---------------- BN+ReLU + Linear(32->2) + two scalar heads -------------------------
__global__ __launch_bounds__(256) void head_kernel(
    const float* __restrict__ H, const float* __restrict__ scale, const float* __restrict__ shift,
    const float* __restrict__ w2, const float* __restrict__ b2,
    const float* __restrict__ hw0, const float* __restrict__ hb0,
    const float* __restrict__ hw1, const float* __restrict__ hb1,
    float* __restrict__ out, int mode) {
  int i = blockIdx.x * 256 + threadIdx.x;
  if (i >= B_N) return;
  const float* row = H + (size_t)i * 32;
  float o0 = b2[0], o1 = b2[1];
#pragma unroll
  for (int k = 0; k < 32; ++k) {
    float v = fmaxf(scale[k] * row[k] + shift[k], 0.f);
    o0 += w2[k] * v;
    o1 += w2[32 + k] * v;
  }
  float l0 = hw0[0] * o0 + hw0[1] * o1 + hb0[0];
  float l1 = hw1[0] * o0 + hw1[1] * o1 + hb1[0];
  float* o = out + (size_t)i * 7;
  if (mode == 0) { o[0] = l0; o[1] = l1; o[2] = sigmoidf_(l0); }
  else           { o[3] = l0; o[4] = l1; o[5] = sigmoidf_(l0); o[6] = sigmoidf_(l1); }
}

// =====================================================================================
extern "C" void kernel_launch(void* const* d_in, const int* in_sizes, int n_in,
                              void* d_out, int out_size, void* d_ws, size_t ws_size,
                              hipStream_t stream) {
  (void)in_sizes; (void)n_in; (void)out_size; (void)ws_size;
  // d_in order: x_input, rebate, then params in dict-insertion order.
  const int*   xin    = (const int*)d_in[0];
  const float* rebate = (const float*)d_in[1];
  const float* emb  = (const float*)d_in[2];
  const float* Qw = (const float*)d_in[3],  *Qb = (const float*)d_in[4];
  const float* Kw = (const float*)d_in[5],  *Kb = (const float*)d_in[6];
  const float* Vw = (const float*)d_in[7],  *Vb = (const float*)d_in[8];
  const float* a1w = (const float*)d_in[9];
  const float* a2w = (const float*)d_in[10], *a2b = (const float*)d_in[11];
  const float* a3w = (const float*)d_in[12];
  const float* tw  = (const float*)d_in[13], *tb  = (const float*)d_in[14];
  const float* c_w0 = (const float*)d_in[15], *c_b0 = (const float*)d_in[16];
  const float* c_g0 = (const float*)d_in[17], *c_be0 = (const float*)d_in[18];
  const float* c_w1 = (const float*)d_in[19], *c_b1 = (const float*)d_in[20];
  const float* c_g1 = (const float*)d_in[21], *c_be1 = (const float*)d_in[22];
  const float* c_w2 = (const float*)d_in[23], *c_b2 = (const float*)d_in[24];
  const float* u_w0 = (const float*)d_in[25], *u_b0 = (const float*)d_in[26];
  const float* u_g0 = (const float*)d_in[27], *u_be0 = (const float*)d_in[28];
  const float* u_w1 = (const float*)d_in[29], *u_b1 = (const float*)d_in[30];
  const float* u_g1 = (const float*)d_in[31], *u_be1 = (const float*)d_in[32];
  const float* u_w2 = (const float*)d_in[33], *u_b2 = (const float*)d_in[34];
  const float* cl_w = (const float*)d_in[35], *cl_b = (const float*)d_in[36];
  const float* ct_w = (const float*)d_in[37], *ct_b = (const float*)d_in[38];
  const float* tl_w = (const float*)d_in[39], *tl_b = (const float*)d_in[40];
  const float* ut_w = (const float*)d_in[41], *ut_b = (const float*)d_in[42];
  float* out = (float*)d_out;

  // workspace layout (bytes, all 16B-aligned)
  char* ws = (char*)d_ws;
  size_t off = 0;
  u16*   flat  = (u16*)(ws + off);  off += (size_t)B_N * 3200 * 2;   // 104.9 MB
  u16*   tfeat = (u16*)(ws + off);  off += (size_t)B_N * 64 * 2;
  float* hc0   = (float*)(ws + off); off += (size_t)B_N * 64 * 4;
  u16*   hc0n  = (u16*)(ws + off);  off += (size_t)B_N * 64 * 2;
  float* hc1   = (float*)(ws + off); off += (size_t)B_N * 32 * 4;
  float* hu0   = (float*)(ws + off); off += (size_t)B_N * 64 * 4;
  u16*   hu0n  = (u16*)(ws + off);  off += (size_t)B_N * 64 * 2;
  float* hu1   = (float*)(ws + off); off += (size_t)B_N * 32 * 4;
  u16* qwb  = (u16*)(ws + off); off += 4096 * 2;
  u16* kwb  = (u16*)(ws + off); off += 4096 * 2;
  u16* vwb  = (u16*)(ws + off); off += 4096 * 2;
  u16* cw0b = (u16*)(ws + off); off += (size_t)64 * 3200 * 2;
  u16* cw1b = (u16*)(ws + off); off += 32 * 64 * 2;
  u16* uw0b = (u16*)(ws + off); off += 64 * 64 * 2;
  u16* uw1b = (u16*)(ws + off); off += 32 * 64 * 2;
  float* st_c0 = (float*)(ws + off); off += 512;   // scale[64] | shift[64]
  float* st_c1 = (float*)(ws + off); off += 512;
  float* st_u0 = (float*)(ws + off); off += 512;
  float* st_u1 = (float*)(ws + off); off += 512;

  // 1) weights -> bf16
  cvt_bf16_kernel<<<16, 256, 0, stream>>>(Qw, qwb, 4096);
  cvt_bf16_kernel<<<16, 256, 0, stream>>>(Kw, kwb, 4096);
  cvt_bf16_kernel<<<16, 256, 0, stream>>>(Vw, vwb, 4096);
  cvt_bf16_kernel<<<800, 256, 0, stream>>>(c_w0, cw0b, 64 * 3200);
  cvt_bf16_kernel<<<8, 256, 0, stream>>>(c_w1, cw1b, 32 * 64);
  cvt_bf16_kernel<<<16, 256, 0, stream>>>(u_w0, uw0b, 64 * 64);
  cvt_bf16_kernel<<<8, 256, 0, stream>>>(u_w1, uw1b, 32 * 64);

  // 2) fused attention + treatment interaction (per-sample workgroups)
  attn_kernel<<<B_N, 256, 0, stream>>>(xin, rebate, emb, Qb, Kb, Vb, qwb, kwb, vwb,
                                       a1w, a2w, a2b, a3w, tw, tb, flat, tfeat);

  // 3) control branch: 3200 -> 64 -> 32 -> heads
  gemm_bf16_kernel<<<B_N / 64, 256, 0, stream>>>(flat, cw0b, c_b0, hc0, 3200, 64);
  bnstats_kernel<<<64, 256, 0, stream>>>(hc0, B_N, 64, c_g0, c_be0, st_c0, st_c0 + 64);
  bnrelu_bf16_kernel<<<(B_N * 64 + 255) / 256, 256, 0, stream>>>(hc0, st_c0, st_c0 + 64, hc0n, B_N * 64, 63);
  gemm_bf16_kernel<<<B_N / 64, 256, 0, stream>>>(hc0n, cw1b, c_b1, hc1, 64, 32);
  bnstats_kernel<<<32, 256, 0, stream>>>(hc1, B_N, 32, c_g1, c_be1, st_c1, st_c1 + 32);
  head_kernel<<<B_N / 256, 256, 0, stream>>>(hc1, st_c1, st_c1 + 32, c_w2, c_b2,
                                             cl_w, cl_b, ct_w, ct_b, out, 0);

  // 4) treatment branch: 64 -> 64 -> 32 -> heads
  gemm_bf16_kernel<<<B_N / 64, 256, 0, stream>>>(tfeat, uw0b, u_b0, hu0, 64, 64);
  bnstats_kernel<<<64, 256, 0, stream>>>(hu0, B_N, 64, u_g0, u_be0, st_u0, st_u0 + 64);
  bnrelu_bf16_kernel<<<(B_N * 64 + 255) / 256, 256, 0, stream>>>(hu0, st_u0, st_u0 + 64, hu0n, B_N * 64, 63);
  gemm_bf16_kernel<<<B_N / 64, 256, 0, stream>>>(hu0n, uw1b, u_b1, hu1, 64, 32);
  bnstats_kernel<<<32, 256, 0, stream>>>(hu1, B_N, 32, u_g1, u_be1, st_u1, st_u1 + 32);
  head_kernel<<<B_N / 256, 256, 0, stream>>>(hu1, st_u1, st_u1 + 32, u_w2, u_b2,
                                             tl_w, tl_b, ut_w, ut_b, out, 1);
}
